// GNN_69114613727581
// MI455X (gfx1250) — matrix operand order repaired and checked by reading, not compiled, not run on hardware
//
#include <hip/hip_runtime.h>

#define NNODES 100000
#define NEDGES 1600000
#define DIM 128
#define NGRAPHS 64

typedef __attribute__((ext_vector_type(16))) __bf16 v16bf;
typedef __attribute__((ext_vector_type(8)))  __bf16 v8bf;
typedef __attribute__((ext_vector_type(8)))  float  v8f;

// ---------------------------------------------------------------- utilities

__global__ __launch_bounds__(256) void zero_kernel(float* __restrict__ p, int n) {
    int i = blockIdx.x * 256 + threadIdx.x;
    if (i < n) p[i] = 0.0f;
}

// One wave32 per edge: lane handles 4 contiguous floats (32*4 = 128 = DIM).
// x[src] rows live in L2 (51MB feature matrix << 192MB L2); adds via f32 atomics.
__global__ __launch_bounds__(256) void scatter_kernel(const float* __restrict__ hin,
                                                      const int*   __restrict__ ei,
                                                      float*       __restrict__ agg) {
    const int e = blockIdx.x * 8 + (threadIdx.x >> 5);
    if (e >= NEDGES) return;
    const int lane = threadIdx.x & 31;
    const int src = ei[e];
    const int dst = ei[NEDGES + e];
    const float4 v = *(const float4*)(hin + (size_t)src * DIM + lane * 4);
    float* o = agg + (size_t)dst * DIM + lane * 4;
    atomicAdd(o + 0, v.x);
    atomicAdd(o + 1, v.y);
    atomicAdd(o + 2, v.z);
    atomicAdd(o + 3, v.w);
}

// ------------------------------------------------- fused GIN MLP (WMMA bf16)
// Computes the transposed product D = W^T x X^T so that:
//   A-frag (weights)  : 2 x ds_load_b128 from transposed LDS weights
//   B-frag (features) : contiguous vector loads (global f32 / LDS bf16),
//                       shared across all 8 N-tiles of a k-chunk
//   D layout          : lane = node row -> vectorized epilogue stores
__global__ __launch_bounds__(256) void gin_mlp_kernel(const float* __restrict__ hin,
                                                      const float* __restrict__ agg,
                                                      const float* __restrict__ w1,
                                                      const float* __restrict__ b1,
                                                      const float* __restrict__ w2,
                                                      const float* __restrict__ b2,
                                                      float* __restrict__ hout,
                                                      int doRelu) {
    __shared__ __bf16 ldsW[DIM * DIM];   // 32 KB: W^T (ldsW[n*128+k]); W1 then W2
    __shared__ __bf16 ldsH[DIM * DIM];   // 32 KB: relu intermediate, row-major [m][n1]

    const int tid  = threadIdx.x;
    const int lane = tid & 31;
    const int wave = tid >> 5;
    const int half = lane >> 4;          // 0: lanes 0-15, 1: lanes 16-31
    const int l15  = lane & 15;
    const int rowStrip = blockIdx.x * 128 + wave * 16;

    // ---- stage W1^T -> LDS (f32 -> bf16); lane-coalesced global, b128 LDS stores
    for (int c = tid; c < DIM * (DIM / 8); c += 256) {
        const int n  = c & (DIM - 1);
        const int k0 = (c >> 7) * 8;
        v8bf pk;
        #pragma unroll
        for (int i = 0; i < 8; ++i) pk[i] = (__bf16)w1[(k0 + i) * DIM + n];
        *(v8bf*)&ldsW[n * DIM + k0] = pk;
    }
    __syncthreads();

    v8f acc[8] = {};

    const int    rowB  = rowStrip + l15;        // node row owned by this lane
    const bool   okB   = rowB < NNODES;
    const size_t baseB = (size_t)rowB * DIM;

    // ---------------- GEMM 1: D1[n1][m] = sum_k W1[k][n1] * (agg+hin)[m][k]
    #pragma unroll
    for (int kk = 0; kk < 4; ++kk) {
        const int k0    = kk * 32;
        const int kbase = k0 + half * 16;   // B layout: 16 consecutive K per lane
        v16bf bN;
        if (okB) {
            const float4* pa = (const float4*)(agg + baseB + kbase);
            const float4* ph = (const float4*)(hin + baseB + kbase);
            #pragma unroll
            for (int q = 0; q < 4; ++q) {
                const float4 va = pa[q];
                const float4 vh = ph[q];
                bN[q * 4 + 0] = (__bf16)(va.x + vh.x);
                bN[q * 4 + 1] = (__bf16)(va.y + vh.y);
                bN[q * 4 + 2] = (__bf16)(va.z + vh.z);
                bN[q * 4 + 3] = (__bf16)(va.w + vh.w);
            }
        } else {
            #pragma unroll
            for (int i = 0; i < 16; ++i) bN[i] = (__bf16)0.0f;
        }
        const int klo = k0 + half * 16;     // A layout: 16 consecutive K per lane (K=klo+i)
        #pragma unroll
        for (int t = 0; t < 8; ++t) {
            const int n = t * 16 + l15;
            const v8bf alo = *(const v8bf*)&ldsW[n * DIM + klo];
            const v8bf ahi = *(const v8bf*)&ldsW[n * DIM + klo + 8];
            const v16bf aW = __builtin_shufflevector(alo, ahi,
                0, 1, 2, 3, 4, 5, 6, 7, 8, 9, 10, 11, 12, 13, 14, 15);
            acc[t] = __builtin_amdgcn_wmma_f32_16x16x32_bf16(
                false, aW, false, bN, (short)0, acc[t], false, false);
        }
    }

    // epilogue 1: lane owns node row mloc, columns n1 = t*16+half*8+r contiguous
    const int mloc = wave * 16 + l15;
    #pragma unroll
    for (int t = 0; t < 8; ++t) {
        const int nb = t * 16 + half * 8;
        const float4 bb0 = *(const float4*)(b1 + nb);
        const float4 bb1 = *(const float4*)(b1 + nb + 4);
        v8bf pk;
        pk[0] = (__bf16)fmaxf(acc[t][0] + bb0.x, 0.0f);
        pk[1] = (__bf16)fmaxf(acc[t][1] + bb0.y, 0.0f);
        pk[2] = (__bf16)fmaxf(acc[t][2] + bb0.z, 0.0f);
        pk[3] = (__bf16)fmaxf(acc[t][3] + bb0.w, 0.0f);
        pk[4] = (__bf16)fmaxf(acc[t][4] + bb1.x, 0.0f);
        pk[5] = (__bf16)fmaxf(acc[t][5] + bb1.y, 0.0f);
        pk[6] = (__bf16)fmaxf(acc[t][6] + bb1.z, 0.0f);
        pk[7] = (__bf16)fmaxf(acc[t][7] + bb1.w, 0.0f);
        *(v8bf*)&ldsH[mloc * DIM + nb] = pk;
    }

    __syncthreads();
    // ---- re-stage W2^T over W1^T (keeps static LDS at 64 KB)
    for (int c = tid; c < DIM * (DIM / 8); c += 256) {
        const int n  = c & (DIM - 1);
        const int k0 = (c >> 7) * 8;
        v8bf pk;
        #pragma unroll
        for (int i = 0; i < 8; ++i) pk[i] = (__bf16)w2[(k0 + i) * DIM + n];
        *(v8bf*)&ldsW[n * DIM + k0] = pk;
    }
    __syncthreads();

    {
        const v8f z = {};
        #pragma unroll
        for (int t = 0; t < 8; ++t) acc[t] = z;
    }

    // ---------------- GEMM 2: D2[n2][m] = sum_n1 W2[n1][n2] * H[m][n1]
    #pragma unroll
    for (int kk = 0; kk < 4; ++kk) {
        const int k0    = kk * 32;
        const int kbase = k0 + half * 16;
        const v16bf bH = *(const v16bf*)&ldsH[mloc * DIM + kbase];  // 32B contiguous
        const int klo = k0 + half * 16;
        #pragma unroll
        for (int t = 0; t < 8; ++t) {
            const int n = t * 16 + l15;
            const v8bf alo = *(const v8bf*)&ldsW[n * DIM + klo];
            const v8bf ahi = *(const v8bf*)&ldsW[n * DIM + klo + 8];
            const v16bf aW = __builtin_shufflevector(alo, ahi,
                0, 1, 2, 3, 4, 5, 6, 7, 8, 9, 10, 11, 12, 13, 14, 15);
            acc[t] = __builtin_amdgcn_wmma_f32_16x16x32_bf16(
                false, aW, false, bH, (short)0, acc[t], false, false);
        }
    }

    // epilogue 2: lane owns one output row; 2 x b128 global stores per tile
    if (okB) {
        float* orow = hout + baseB;
        #pragma unroll
        for (int t = 0; t < 8; ++t) {
            const int nb = t * 16 + half * 8;
            const float4 bb0 = *(const float4*)(b2 + nb);
            const float4 bb1 = *(const float4*)(b2 + nb + 4);
            float4 o0, o1;
            o0.x = acc[t][0] + bb0.x;  o0.y = acc[t][1] + bb0.y;
            o0.z = acc[t][2] + bb0.z;  o0.w = acc[t][3] + bb0.w;
            o1.x = acc[t][4] + bb1.x;  o1.y = acc[t][5] + bb1.y;
            o1.z = acc[t][6] + bb1.z;  o1.w = acc[t][7] + bb1.w;
            if (doRelu) {
                o0.x = fmaxf(o0.x, 0.0f); o0.y = fmaxf(o0.y, 0.0f);
                o0.z = fmaxf(o0.z, 0.0f); o0.w = fmaxf(o0.w, 0.0f);
                o1.x = fmaxf(o1.x, 0.0f); o1.y = fmaxf(o1.y, 0.0f);
                o1.z = fmaxf(o1.z, 0.0f); o1.w = fmaxf(o1.w, 0.0f);
            }
            *(float4*)(orow + nb)     = o0;
            *(float4*)(orow + nb + 4) = o1;
        }
    }
}

// ---------------------------------------------------------------- pooling

__global__ __launch_bounds__(256) void pool_sum_kernel(const float* __restrict__ h,
                                                       const int*   __restrict__ batch,
                                                       float* __restrict__ sums,
                                                       float* __restrict__ cnts) {
    const int node = blockIdx.x * 8 + (threadIdx.x >> 5);
    if (node >= NNODES) return;
    const int lane = threadIdx.x & 31;
    const int g = batch[node];
    const float4 v = *(const float4*)(h + (size_t)node * DIM + lane * 4);
    float* o = sums + (size_t)g * DIM + lane * 4;
    atomicAdd(o + 0, v.x);
    atomicAdd(o + 1, v.y);
    atomicAdd(o + 2, v.z);
    atomicAdd(o + 3, v.w);
    if (lane == 0) atomicAdd(cnts + g, 1.0f);
}

__global__ __launch_bounds__(256) void pool_div_kernel(const float* __restrict__ sums,
                                                       const float* __restrict__ cnts,
                                                       float* __restrict__ out) {
    const int i = blockIdx.x * 256 + threadIdx.x;
    if (i < NGRAPHS * DIM) out[i] = sums[i] / fmaxf(cnts[i >> 7], 1.0f);
}

// ---------------------------------------------------------------- launch

extern "C" void kernel_launch(void* const* d_in, const int* in_sizes, int n_in,
                              void* d_out, int out_size, void* d_ws, size_t ws_size,
                              hipStream_t stream) {
    const float* x     = (const float*)d_in[0];
    const int*   ei    = (const int*)d_in[1];
    const int*   batch = (const int*)d_in[2];
    const float* w1_1 = (const float*)d_in[3];
    const float* b1_1 = (const float*)d_in[4];
    const float* w2_1 = (const float*)d_in[5];
    const float* b2_1 = (const float*)d_in[6];
    const float* w1_2 = (const float*)d_in[7];
    const float* b1_2 = (const float*)d_in[8];
    const float* w2_2 = (const float*)d_in[9];
    const float* b2_2 = (const float*)d_in[10];
    const float* w1_3 = (const float*)d_in[11];
    const float* b1_3 = (const float*)d_in[12];
    const float* w2_3 = (const float*)d_in[13];
    const float* b2_3 = (const float*)d_in[14];

    const size_t feat = (size_t)NNODES * DIM;     // 12.8M floats
    float* agg  = (float*)d_ws;
    float* hA   = agg + feat;
    float* hB   = hA + feat;
    float* sums = hB + feat;
    float* cnts = sums + NGRAPHS * DIM;

    const int featN = (int)feat;
    dim3 blk(256);
    dim3 zgrid((featN + 255) / 256);
    dim3 sgrid((NEDGES + 7) / 8);
    dim3 mgrid((NNODES + 127) / 128);
    dim3 pgrid((NNODES + 7) / 8);

    // layer 1
    zero_kernel<<<zgrid, blk, 0, stream>>>(agg, featN);
    scatter_kernel<<<sgrid, blk, 0, stream>>>(x, ei, agg);
    gin_mlp_kernel<<<mgrid, blk, 0, stream>>>(x, agg, w1_1, b1_1, w2_1, b2_1, hA, 1);
    // layer 2
    zero_kernel<<<zgrid, blk, 0, stream>>>(agg, featN);
    scatter_kernel<<<sgrid, blk, 0, stream>>>(hA, ei, agg);
    gin_mlp_kernel<<<mgrid, blk, 0, stream>>>(hA, agg, w1_2, b1_2, w2_2, b2_2, hB, 1);
    // layer 3 (no final relu)
    zero_kernel<<<zgrid, blk, 0, stream>>>(agg, featN);
    scatter_kernel<<<sgrid, blk, 0, stream>>>(hB, ei, agg);
    gin_mlp_kernel<<<mgrid, blk, 0, stream>>>(hB, agg, w1_3, b1_3, w2_3, b2_3, hA, 0);
    // mean pool
    zero_kernel<<<dim3((NGRAPHS * DIM + NGRAPHS + 255) / 256), blk, 0, stream>>>(
        sums, NGRAPHS * DIM + NGRAPHS);
    pool_sum_kernel<<<pgrid, blk, 0, stream>>>(hA, batch, sums, cnts);
    pool_div_kernel<<<dim3((NGRAPHS * DIM + 255) / 256), blk, 0, stream>>>(
        sums, cnts, (float*)d_out);
}
